// GraphModule_15496242004314
// MI455X (gfx1250) — matrix operand order repaired and checked by reading, not compiled, not run on hardware
//
#include <hip/hip_runtime.h>
#include <hip/hip_bf16.h>

// ---------------------------------------------------------------------------
// MI455X (gfx1250) implementation.
//
// Key math: g_rep / feat_end / feat_start have identical rows per batch, so
// both cross-attentions degenerate to uniform softmax => single linear maps.
// _bn_cl on (B, N*N, D) takes per-row stats over (B,D); with the rank-
// structured msg = Vix[i] + Vjx[j] + Ee, those stats have a closed form from
// per-i/per-j sums and a 27x27 cross-dot. All dense GEMMs run through
// v_wmma_f32_16x16x32_bf16 (bf16 in, f32 accumulate).
// Workspace use: ~25.3 MB.
// ---------------------------------------------------------------------------

typedef __attribute__((ext_vector_type(16))) __bf16 v16bf;
typedef __attribute__((ext_vector_type(8)))  float  v8f;

#define B_   64
#define N_   27
#define D_   512
#define BD_  (B_ * D_)     // 32768
#define ROWS_ (B_ * N_)    // 1728
#define NN_  (N_ * N_)     // 729
#define EPS_ 1e-5f

// stat-region float indices
#define SA_   0
#define SA2_  27
#define SP_   54
#define SC_   81
#define SC2_  108
#define SQ_   135
#define SE_   162
#define SE2_  163
#define SM1_  164          // 729 entries
#define SMU_  896          // 729 entries
#define SRS_  1632         // 729 entries
#define SBNM_ 2368         // 27 entries
#define SBNR_ 2400         // 27 entries
#define SENM_ 2432         // mu, rstd for edge-BN scalars

union FragBF { uint4 q[2]; v16bf v; };

// --------------------------- WMMA GEMM -------------------------------------
// C[M,Nt] = A[M,K](bf16) * W[Nt,K](bf16)^T + bias, fp32 accumulate.
// One wave -> 16x64 tile (4 accumulators). M%16==0, Nt%64==0, K%32==0.
__global__ __launch_bounds__(256) void gemm_bf16_wmma(
    const __hip_bfloat16* __restrict__ A, const __hip_bfloat16* __restrict__ W,
    const float* __restrict__ bias, float* __restrict__ C,
    int M, int Nt, int K)
{
    const int tilesN = Nt >> 6;
    const int total  = (M >> 4) * tilesN;
    const int wid    = blockIdx.x * 8 + (threadIdx.x >> 5);
    if (wid >= total) return;                 // wave-uniform: EXEC all-1s inside
    const int tm   = wid / tilesN;
    const int tn   = wid - tm * tilesN;
    const int lane = threadIdx.x & 31;
    const int r    = lane & 15;
    const int hi   = lane >> 4;               // lane half selects K sub-chunks

    const __hip_bfloat16* Arow = A + (size_t)(tm * 16 + r) * K;
    const __hip_bfloat16* Wrow = W + (size_t)(tn * 64 + r) * K;

    v8f c0 = {}, c1 = {}, c2 = {}, c3 = {};
    for (int kk = 0; kk < K; kk += 32) {
        FragBF a, b0, b1, b2, b3;
        // A 16x32 bf16 frag: lane half selects 16 contiguous K values of row r
        const __hip_bfloat16* ap = Arow + kk + hi * 16;
        a.q[0] = *(const uint4*)(ap);
        a.q[1] = *(const uint4*)(ap + 8);
        // B 32x16 frag: lane half selects 16 contiguous K values of column r
        const __hip_bfloat16* wp = Wrow + kk + hi * 16;
        b0.q[0] = *(const uint4*)(wp);
        b0.q[1] = *(const uint4*)(wp + 8);
        b1.q[0] = *(const uint4*)(wp + (size_t)16 * K);
        b1.q[1] = *(const uint4*)(wp + (size_t)16 * K + 8);
        b2.q[0] = *(const uint4*)(wp + (size_t)32 * K);
        b2.q[1] = *(const uint4*)(wp + (size_t)32 * K + 8);
        b3.q[0] = *(const uint4*)(wp + (size_t)48 * K);
        b3.q[1] = *(const uint4*)(wp + (size_t)48 * K + 8);
        c0 = __builtin_amdgcn_wmma_f32_16x16x32_bf16(false, a.v, false, b0.v, (short)0, c0, false, false);
        c1 = __builtin_amdgcn_wmma_f32_16x16x32_bf16(false, a.v, false, b1.v, (short)0, c1, false, false);
        c2 = __builtin_amdgcn_wmma_f32_16x16x32_bf16(false, a.v, false, b2.v, (short)0, c2, false, false);
        c3 = __builtin_amdgcn_wmma_f32_16x16x32_bf16(false, a.v, false, b3.v, (short)0, c3, false, false);
    }
    // C/D layout: vgpr e, lanes 0-15 -> M=e, lanes 16-31 -> M=e+8, N=lane&15
    const int rb = tm * 16 + hi * 8;
    #pragma unroll
    for (int t = 0; t < 4; ++t) {
        v8f c = (t == 0) ? c0 : (t == 1) ? c1 : (t == 2) ? c2 : c3;
        const int cn = tn * 64 + t * 16 + r;
        const float bv = bias ? bias[cn] : 0.0f;
        #pragma unroll
        for (int e = 0; e < 8; ++e)
            C[(size_t)(rb + e) * Nt + cn] = c[e] + bv;
    }
}

// ------------------------- small helper kernels ----------------------------
__global__ void f32_to_bf16(const float* __restrict__ s,
                            __hip_bfloat16* __restrict__ d, int n)
{
    int i = blockIdx.x * 256 + threadIdx.x;
    if (i < n) d[i] = __float2bfloat16(s[i]);
}

// gf[b,d] = mean_n x[b,n,d]  (stored directly as bf16 GEMM input)
__global__ void mean_nodes_bf16(const float* __restrict__ x,
                                __hip_bfloat16* __restrict__ gf)
{
    int i = blockIdx.x * 256 + threadIdx.x;
    if (i >= BD_) return;
    int b = i >> 9, d = i & 511;
    float s = 0.f;
    #pragma unroll
    for (int n = 0; n < N_; ++n) s += x[((size_t)(b * N_ + n) << 9) + d];
    gf[i] = __float2bfloat16(s * (1.0f / N_));
}

// per-channel BN over batch + relu (g path), write bf16 for next GEMM
__global__ void bn_batch_relu_bf16(const float* __restrict__ g0,
                                   __hip_bfloat16* __restrict__ gbf)
{
    int d = blockIdx.x * 256 + threadIdx.x;
    if (d >= D_) return;
    float s1 = 0.f, s2 = 0.f;
    for (int b = 0; b < B_; ++b) { float v = g0[b * D_ + d]; s1 += v; s2 += v * v; }
    float m  = s1 / B_;
    float rs = rsqrtf(s2 / B_ - m * m + EPS_);
    for (int b = 0; b < B_; ++b) {
        float v = (g0[b * D_ + d] - m) * rs;
        gbf[b * D_ + d] = __float2bfloat16(fmaxf(v, 0.0f));
    }
}

// scalar mean/rstd over all BD_ elements (edge BN collapses to scalars)
__global__ void scalar_stats(const float* __restrict__ v, float* __restrict__ o2)
{
    __shared__ float h1[256], h2[256];
    float s1 = 0.f, s2 = 0.f;
    for (int i = threadIdx.x; i < BD_; i += 256) { float a = v[i]; s1 += a; s2 += a * a; }
    h1[threadIdx.x] = s1; h2[threadIdx.x] = s2; __syncthreads();
    for (int o = 128; o > 0; o >>= 1) {
        if (threadIdx.x < o) { h1[threadIdx.x] += h1[threadIdx.x + o];
                               h2[threadIdx.x] += h2[threadIdx.x + o]; }
        __syncthreads();
    }
    if (threadIdx.x == 0) {
        float m = h1[0] / BD_;
        o2[0] = m;
        o2[1] = rsqrtf(h2[0] / BD_ - m * m + EPS_);
    }
}

__global__ void apply_norm(const float* __restrict__ v, const float* __restrict__ st,
                           float* __restrict__ of, __hip_bfloat16* __restrict__ ob)
{
    int i = blockIdx.x * 256 + threadIdx.x;
    if (i >= BD_) return;
    float y = (v[i] - st[0]) * st[1];
    of[i] = y;
    ob[i] = __float2bfloat16(y);
}

// per-i sums of Vix (roles 0..26), per-j sums of Vjx (27..53), Ee sums (54)
__global__ void edge_msg_stats(const float* __restrict__ big4,
                               const float* __restrict__ Ee,
                               float* __restrict__ st)
{
    __shared__ float h0[256], h1[256], h2[256];
    int role = blockIdx.x;
    float s1 = 0.f, s2 = 0.f, s3 = 0.f;
    if (role < 54) {
        int idx = (role < 27) ? role : role - 27;
        int col = (role < 27) ? 0 : 512;
        for (int t = threadIdx.x; t < BD_; t += 256) {
            int b = t >> 9, d = t & 511;
            float a = big4[((size_t)(b * N_ + idx) << 11) + col + d];
            float e = Ee[t];
            s1 += a; s2 += a * a; s3 += a * e;
        }
    } else {
        for (int t = threadIdx.x; t < BD_; t += 256) { float e = Ee[t]; s1 += e; s2 += e * e; }
    }
    h0[threadIdx.x] = s1; h1[threadIdx.x] = s2; h2[threadIdx.x] = s3; __syncthreads();
    for (int o = 128; o > 0; o >>= 1) {
        if (threadIdx.x < o) { h0[threadIdx.x] += h0[threadIdx.x + o];
                               h1[threadIdx.x] += h1[threadIdx.x + o];
                               h2[threadIdx.x] += h2[threadIdx.x + o]; }
        __syncthreads();
    }
    if (threadIdx.x == 0) {
        if (role < 27)      { st[SA_ + role] = h0[0]; st[SA2_ + role] = h1[0]; st[SP_ + role] = h2[0]; }
        else if (role < 54) { int j = role - 27;
                              st[SC_ + j] = h0[0]; st[SC2_ + j] = h1[0]; st[SQ_ + j] = h2[0]; }
        else                { st[SE_] = h0[0]; st[SE2_] = h1[0]; }
    }
}

// M1[i,j] = sum_{b,d} Vix[b,i,d] * Vjx[b,j,d]   (729 blocks)
__global__ void cross_dot(const float* __restrict__ big4, float* __restrict__ st)
{
    __shared__ float h[256];
    int i = blockIdx.x / N_, j = blockIdx.x % N_;
    float s = 0.f;
    for (int t = threadIdx.x; t < BD_; t += 256) {
        int b = t >> 9, d = t & 511;
        s += big4[((size_t)(b * N_ + i) << 11) + d] *
             big4[((size_t)(b * N_ + j) << 11) + 512 + d];
    }
    h[threadIdx.x] = s; __syncthreads();
    for (int o = 128; o > 0; o >>= 1) {
        if (threadIdx.x < o) h[threadIdx.x] += h[threadIdx.x + o];
        __syncthreads();
    }
    if (threadIdx.x == 0) st[SM1_ + blockIdx.x] = h[0];
}

// closed-form per-(i,j) msg-BN mean / rstd
__global__ void finalize_msg_bn(float* __restrict__ st)
{
    int t = threadIdx.x;
    if (t >= NN_) return;
    int i = t / N_, j = t % N_;
    float m  = (st[SA_ + i] + st[SC_ + j] + st[SE_]) * (1.0f / BD_);
    float ss = st[SA2_ + i] + st[SC2_ + j] + st[SE2_]
             + 2.0f * (st[SM1_ + t] + st[SP_ + i] + st[SQ_ + j]);
    st[SMU_ + t] = m;
    st[SRS_ + t] = rsqrtf(ss * (1.0f / BD_) - m * m + EPS_);
}

// fused: msg-BN -> relu -> +edge -> sigmoid -> softmax_j -> dot Ujx -> +U1x
__global__ __launch_bounds__(256) void gcn_fused(
    const float* __restrict__ big4, const float* __restrict__ Ee,
    const float* __restrict__ en,   const float* __restrict__ st,
    float* __restrict__ xn)
{
    int blk   = blockIdx.x;
    int chunk = blk & 1;
    int bi    = blk >> 1;                 // b*N_ + i
    int i     = bi % N_;
    int b     = bi / N_;
    int d     = chunk * 256 + threadIdx.x;
    size_t rowbase = (size_t)bi << 11;
    float a  = big4[rowbase + d];              // Vix[b,i,d]
    float u1 = big4[rowbase + 1536 + d];       // U1x[b,i,d]
    float ee = Ee[b * D_ + d];
    float ev = en[b * D_ + d];
    float s[N_];
    float mx = -1e30f;
    #pragma unroll
    for (int j = 0; j < N_; ++j) {
        float c  = big4[((size_t)(b * N_ + j) << 11) + 512 + d];  // Vjx[b,j,d]
        float t  = a + c + ee;
        float bn = (t - st[SMU_ + i * N_ + j]) * st[SRS_ + i * N_ + j];
        float e2 = ev + fmaxf(bn, 0.0f);
        float sg = 1.0f / (1.0f + __expf(-e2));
        s[j] = sg;
        mx = fmaxf(mx, sg);
    }
    float num = 0.f, den = 0.f;
    #pragma unroll
    for (int j = 0; j < N_; ++j) {
        float w = __expf(s[j] - mx);
        den += w;
        num += w * big4[((size_t)(b * N_ + j) << 11) + 1024 + d]; // Ujx[b,j,d]
    }
    xn[(size_t)bi * D_ + d] = u1 + num / (den * (float)N_);
}

// per-node scalar BN stats of xn over (b,d)
__global__ void node_bn_stats(const float* __restrict__ xn, float* __restrict__ st)
{
    __shared__ float h1[256], h2[256];
    int n = blockIdx.x;
    float s1 = 0.f, s2 = 0.f;
    for (int t = threadIdx.x; t < BD_; t += 256) {
        int b = t >> 9, d = t & 511;
        float v = xn[(size_t)(b * N_ + n) * D_ + d];
        s1 += v; s2 += v * v;
    }
    h1[threadIdx.x] = s1; h2[threadIdx.x] = s2; __syncthreads();
    for (int o = 128; o > 0; o >>= 1) {
        if (threadIdx.x < o) { h1[threadIdx.x] += h1[threadIdx.x + o];
                               h2[threadIdx.x] += h2[threadIdx.x + o]; }
        __syncthreads();
    }
    if (threadIdx.x == 0) {
        float m = h1[0] / BD_;
        st[SBNM_ + n] = m;
        st[SBNR_ + n] = rsqrtf(h2[0] / BD_ - m * m + EPS_);
    }
}

__global__ void final_out(const float* __restrict__ x, const float* __restrict__ xn,
                          const float* __restrict__ st, float* __restrict__ out)
{
    int i = blockIdx.x * 256 + threadIdx.x;
    if (i >= ROWS_ * D_) return;
    int n = (i >> 9) % N_;
    float v = x[i] + (xn[i] - st[SBNM_ + n]) * st[SBNR_ + n];
    out[i] = fmaxf(v, 0.0f);
}

// ---------------------------- workspace layout -----------------------------
static constexpr size_t OFF_XBF   = 0;                       // 1728*512 bf16
static constexpr size_t OFF_WCAT  = OFF_XBF   + 1769472;     // A1|B1|V1|U1 bf16
static constexpr size_t OFF_WGL   = OFF_WCAT  + 2097152;
static constexpr size_t OFF_WFV   = OFF_WGL   + 524288;
static constexpr size_t OFF_WAV   = OFF_WFV   + 524288;
static constexpr size_t OFF_WEP   = OFF_WAV   + 524288;
static constexpr size_t OFF_WE1   = OFF_WEP   + 524288;
static constexpr size_t OFF_GFBF  = OFF_WE1   + 524288;
static constexpr size_t OFF_G0    = OFF_GFBF  + 65536;
static constexpr size_t OFF_GBF   = OFF_G0    + 131072;
static constexpr size_t OFF_FEATV = OFF_GBF   + 65536;
static constexpr size_t OFF_FEATB = OFF_FEATV + 131072;
static constexpr size_t OFF_F2V   = OFF_FEATB + 65536;
static constexpr size_t OFF_F2B   = OFF_F2V   + 131072;
static constexpr size_t OFF_EPV   = OFF_F2B   + 65536;
static constexpr size_t OFF_ENV   = OFF_EPV   + 131072;
static constexpr size_t OFF_ENB   = OFF_ENV   + 131072;
static constexpr size_t OFF_EE    = OFF_ENB   + 65536;
static constexpr size_t OFF_ST    = OFF_EE    + 131072;      // 16 KB stats
static constexpr size_t OFF_BIG4  = OFF_ST    + 16384;       // 1728x2048 f32
static constexpr size_t OFF_XN    = OFF_BIG4  + 14155776;    // 1728x512 f32

extern "C" void kernel_launch(void* const* d_in, const int* in_sizes, int n_in,
                              void* d_out, int out_size, void* d_ws, size_t ws_size,
                              hipStream_t stream)
{
    (void)in_sizes; (void)n_in; (void)out_size; (void)ws_size;
    const float* x    = (const float*)d_in[0];
    const float* gl_w = (const float*)d_in[1];
    const float* gl_b = (const float*)d_in[2];
    const float* fv_w = (const float*)d_in[7];
    const float* fv_b = (const float*)d_in[8];
    const float* av_w = (const float*)d_in[13];
    const float* av_b = (const float*)d_in[14];
    const float* ep_w = (const float*)d_in[15];
    const float* ep_b = (const float*)d_in[16];
    const float* U1   = (const float*)d_in[17];
    const float* V1   = (const float*)d_in[18];
    const float* A1   = (const float*)d_in[19];
    const float* B1   = (const float*)d_in[20];
    const float* E1   = (const float*)d_in[21];
    float* out = (float*)d_out;

    char* ws = (char*)d_ws;
    __hip_bfloat16* xbf   = (__hip_bfloat16*)(ws + OFF_XBF);
    __hip_bfloat16* wcat  = (__hip_bfloat16*)(ws + OFF_WCAT);
    __hip_bfloat16* wgl   = (__hip_bfloat16*)(ws + OFF_WGL);
    __hip_bfloat16* wfv   = (__hip_bfloat16*)(ws + OFF_WFV);
    __hip_bfloat16* wav   = (__hip_bfloat16*)(ws + OFF_WAV);
    __hip_bfloat16* wep   = (__hip_bfloat16*)(ws + OFF_WEP);
    __hip_bfloat16* we1   = (__hip_bfloat16*)(ws + OFF_WE1);
    __hip_bfloat16* gfbf  = (__hip_bfloat16*)(ws + OFF_GFBF);
    float*          g0    = (float*)(ws + OFF_G0);
    __hip_bfloat16* gbf   = (__hip_bfloat16*)(ws + OFF_GBF);
    float*          featv = (float*)(ws + OFF_FEATV);
    __hip_bfloat16* featb = (__hip_bfloat16*)(ws + OFF_FEATB);
    float*          f2v   = (float*)(ws + OFF_F2V);
    __hip_bfloat16* f2b   = (__hip_bfloat16*)(ws + OFF_F2B);
    float*          epv   = (float*)(ws + OFF_EPV);
    float*          env   = (float*)(ws + OFF_ENV);
    __hip_bfloat16* enb   = (__hip_bfloat16*)(ws + OFF_ENB);
    float*          Ee    = (float*)(ws + OFF_EE);
    float*          st    = (float*)(ws + OFF_ST);
    float*          big4  = (float*)(ws + OFF_BIG4);
    float*          xn    = (float*)(ws + OFF_XN);

    const int WW = 512 * 512;             // weight element count
    // ---- bf16 staging (x + 9 live weights; fq/fk/aq/ak are provably dead) ----
    f32_to_bf16<<<3456, 256, 0, stream>>>(x, xbf, ROWS_ * D_);
    f32_to_bf16<<<1024, 256, 0, stream>>>(A1, wcat + 0 * (size_t)WW, WW);
    f32_to_bf16<<<1024, 256, 0, stream>>>(B1, wcat + 1 * (size_t)WW, WW);
    f32_to_bf16<<<1024, 256, 0, stream>>>(V1, wcat + 2 * (size_t)WW, WW);
    f32_to_bf16<<<1024, 256, 0, stream>>>(U1, wcat + 3 * (size_t)WW, WW);
    f32_to_bf16<<<1024, 256, 0, stream>>>(gl_w, wgl, WW);
    f32_to_bf16<<<1024, 256, 0, stream>>>(fv_w, wfv, WW);
    f32_to_bf16<<<1024, 256, 0, stream>>>(av_w, wav, WW);
    f32_to_bf16<<<1024, 256, 0, stream>>>(ep_w, wep, WW);
    f32_to_bf16<<<1024, 256, 0, stream>>>(E1,   we1, WW);

    // ---- global path: gf -> g -> BN+relu ----
    mean_nodes_bf16<<<128, 256, 0, stream>>>(x, gfbf);
    gemm_bf16_wmma<<<4, 256, 0, stream>>>(gfbf, wgl, gl_b, g0, B_, D_, D_);
    bn_batch_relu_bf16<<<2, 256, 0, stream>>>(g0, gbf);

    // ---- degenerate attn chain: featv -> feat2v -> epv -> edge scalars ----
    gemm_bf16_wmma<<<4, 256, 0, stream>>>(gbf, wfv, fv_b, featv, B_, D_, D_);
    f32_to_bf16<<<128, 256, 0, stream>>>(featv, featb, BD_);
    gemm_bf16_wmma<<<4, 256, 0, stream>>>(featb, wav, av_b, f2v, B_, D_, D_);
    f32_to_bf16<<<128, 256, 0, stream>>>(f2v, f2b, BD_);
    gemm_bf16_wmma<<<4, 256, 0, stream>>>(f2b, wep, ep_b, epv, B_, D_, D_);
    scalar_stats<<<1, 256, 0, stream>>>(epv, st + SENM_);
    apply_norm<<<128, 256, 0, stream>>>(epv, st + SENM_, env, enb);
    gemm_bf16_wmma<<<4, 256, 0, stream>>>(enb, we1, nullptr, Ee, B_, D_, D_);

    // ---- fused node GEMM: [Vix|Vjx|Ujx|U1x] = x @ [A1|B1|V1|U1]^T ----
    gemm_bf16_wmma<<<432, 256, 0, stream>>>(xbf, wcat, nullptr, big4,
                                            ROWS_, 4 * D_, D_);

    // ---- closed-form msg-BN stats, then fused GCN epilogue ----
    edge_msg_stats<<<55, 256, 0, stream>>>(big4, Ee, st);
    cross_dot<<<NN_, 256, 0, stream>>>(big4, st);
    finalize_msg_bn<<<1, 736, 0, stream>>>(st);
    gcn_fused<<<ROWS_ * 2, 256, 0, stream>>>(big4, Ee, env, st, xn);

    // ---- final per-node BN + residual relu ----
    node_bn_stats<<<N_, 256, 0, stream>>>(xn, st);
    final_out<<<3456, 256, 0, stream>>>(x, xn, st, out);
}
